// HomoAttentionConv_9363028705369
// MI455X (gfx1250) — matrix operand order repaired and checked by reading, compiled, not run on hardware
//
#include <hip/hip_runtime.h>

// -------- CDNA5 (gfx1250) wave32 WMMA types --------
typedef __attribute__((ext_vector_type(16))) _Float16 v16h;
typedef __attribute__((ext_vector_type(8)))  float    v8f;

#define NN 100000   // nodes (16*6250)
#define NE 800000   // edges (16*50000)
#define D  64

// ---------- monotonic float<->uint mapping for atomic segment-max ----------
__device__ __forceinline__ unsigned fmap(float f) {
  int ix = __float_as_int(f);
  return (ix >= 0) ? ((unsigned)ix ^ 0x80000000u) : ~(unsigned)ix;
}
__device__ __forceinline__ float funmap(unsigned u) {
  int ix = (u & 0x80000000u) ? (int)(u ^ 0x80000000u) : (int)(~u);
  return __int_as_float(ix);
}

// ---------- A fragment loader (f32 row-major -> f16 16x32 fragment) ----------
__device__ __forceinline__ v16h load_a_f32(const float* A, int lda, int row0,
                                           int k0, int lane) {
  const float* p = A + (size_t)(row0 + (lane & 15)) * lda + k0 + ((lane >> 4) << 3);
  v16h a;
#pragma unroll
  for (int i = 0; i < 8; ++i) a[i] = (_Float16)p[i];
#pragma unroll
  for (int i = 0; i < 8; ++i) a[8 + i] = (_Float16)p[16 + i];
  return a;
}

// ---------- pre-packed B fragments (global) ----------
__global__ void pack_b_kernel(const float* __restrict__ B, _Float16* __restrict__ Bp,
                              int K, int N) {
  int idx = blockIdx.x * blockDim.x + threadIdx.x;
  if (idx >= K * N) return;
  int i    = idx & 15;
  int lane = (idx >> 4) & 31;
  int frag = idx >> 9;
  int nt   = frag % (N >> 4);
  int k0c  = frag / (N >> 4);
  int k = (k0c << 5) + ((i >> 3) << 4) + ((lane >> 4) << 3) + (i & 7);
  int n = (nt << 4) + (lane & 15);
  Bp[idx] = (_Float16)B[(size_t)k * N + n];
}

// stage packed B into LDS (whole block cooperates), then barrier
__device__ __forceinline__ void stage_b_lds(_Float16* smem, const _Float16* Bp,
                                            int nhalfs) {
  for (int i = threadIdx.x; i < (nhalfs >> 3); i += blockDim.x)
    ((int4*)smem)[i] = ((const int4*)Bp)[i];
  __syncthreads();
}

__device__ __forceinline__ v16h load_b_lds(const _Float16* smem, int k0c, int NT,
                                           int nt, int lane) {
  return ((const v16h*)smem)[(k0c * NT + nt) * 32 + lane];
}

// C/D (16x16 f32): VGPR r -> row = row0 + (lane>=16 ? 8 : 0) + r, col = n0+(lane&15)
template <int ACT>  // 0=none, 1=tanh, 2=relu
__device__ __forceinline__ void store_c(float* C, int ldc, int row0, int n0,
                                        int lane, v8f acc, const float* bias) {
  int col = n0 + (lane & 15);
  int rb  = row0 + ((lane >> 4) << 3);
  float bv = bias ? bias[col] : 0.0f;
#pragma unroll
  for (int r = 0; r < 8; ++r) {
    float v = acc[r] + bv;
    if (ACT == 1) v = tanhf(v);
    if (ACT == 2) v = fmaxf(v, 0.0f);
    C[(size_t)(rb + r) * ldc + col] = v;
  }
}

// one K-step: NT WMMAs with B fragments from LDS
template <int NT>
__device__ __forceinline__ void kstep(v16h a, const _Float16* smem, int k0c,
                                      int lane, v8f (&acc)[NT]) {
#pragma unroll
  for (int nt = 0; nt < NT; ++nt) {
    v16h b = load_b_lds(smem, k0c, NT, nt, lane);
    acc[nt] = __builtin_amdgcn_wmma_f32_16x16x32_f16(false, a, false, b,
                                                     (short)0, acc[nt],
                                                     false, false);
  }
}

// ---------- generic panel GEMM: one wave = 16 rows x (NT*16) cols ----------
template <int ACT, int NT>
__global__ void gemm_kernel(const float* __restrict__ A, int lda,
                            const _Float16* __restrict__ Bp,  // packed K x (NT*16)
                            const float* __restrict__ bias,   // may be null
                            float* __restrict__ C, int ldc,
                            int M, int K) {
  extern __shared__ _Float16 smem[];
  stage_b_lds(smem, Bp, K * NT * 16);
  int lane = threadIdx.x & 31;
  int wave = (blockIdx.x * blockDim.x + threadIdx.x) >> 5;
  int nwav = (gridDim.x * blockDim.x) >> 5;
  int rtiles = M >> 4;
  for (int t = wave; t < rtiles; t += nwav) {
    int row0 = t << 4;
    // prefetch next row-tile of A while this tile computes
    __builtin_prefetch(A + (size_t)(row0 + (nwav << 4) + (lane & 15)) * lda, 0, 1);
    v8f acc[NT];
#pragma unroll
    for (int nt = 0; nt < NT; ++nt) acc[nt] = (v8f){};
#pragma clang loop unroll(disable)
    for (int k0 = 0; k0 < K; k0 += 32) {
      v16h a = load_a_f32(A, lda, row0, k0, lane);
      kstep<NT>(a, smem, k0 >> 5, lane, acc);
    }
#pragma unroll
    for (int nt = 0; nt < NT; ++nt)
      store_c<ACT>(C, ldc, row0, nt << 4, lane, acc[nt], bias);
  }
}

// ---------- init: zero denom/maxu/mixU every call (ws is poisoned) ----------
__global__ void init_kernel(float* __restrict__ denom, unsigned* __restrict__ maxu,
                            float* __restrict__ mixU) {
  int i = blockIdx.x * blockDim.x + threadIdx.x;
  if (i < NN * 128) mixU[i] = 0.0f;
  if (i < NN) { denom[i] = 0.0f; maxu[i] = 0u; }
}

// ---------- pass 1: one wave per edge; score = q[dst] . [x[dst]|efeat] ----------
__global__ void score_kernel(const float* __restrict__ q,
                             const float* __restrict__ x,
                             const float* __restrict__ efeat,
                             const int* __restrict__ dst,
                             float* __restrict__ scores,
                             unsigned* __restrict__ maxu) {
  int lane = threadIdx.x & 31;
  int e    = (blockIdx.x * blockDim.x + threadIdx.x) >> 5;
  if (e >= NE) return;
  int d = dst[e];
  float4 qv = *(const float4*)(q + (size_t)d * 128 + lane * 4);
  float4 hv = (lane < 16)
      ? *(const float4*)(x + (size_t)d * D + lane * 4)
      : *(const float4*)(efeat + (size_t)e * D + (lane - 16) * 4);
  float s = qv.x * hv.x + qv.y * hv.y + qv.z * hv.z + qv.w * hv.w;
#pragma unroll
  for (int off = 16; off; off >>= 1) s += __shfl_xor(s, off, 32);
  if (lane == 0) {
    scores[e] = s;
    atomicMax(maxu + d, fmap(s));
  }
}

// ---------- pass 2: ex = exp(s - max); accumulate denom and unnormalized mix ----------
__global__ void softmax_accum_kernel(const float* __restrict__ scores,
                                     const unsigned* __restrict__ maxu,
                                     const float* __restrict__ x,
                                     const float* __restrict__ efeat,
                                     const int* __restrict__ dst,
                                     float* __restrict__ denom,
                                     float* __restrict__ mixU) {
  int lane = threadIdx.x & 31;
  int e    = (blockIdx.x * blockDim.x + threadIdx.x) >> 5;
  if (e >= NE) return;
  int d = dst[e];
  float ex = __expf(scores[e] - funmap(maxu[d]));
  if (lane == 0) atomicAdd(denom + d, ex);
  float4 hv = (lane < 16)
      ? *(const float4*)(x + (size_t)d * D + lane * 4)
      : *(const float4*)(efeat + (size_t)e * D + (lane - 16) * 4);
  float* m = mixU + (size_t)d * 128 + lane * 4;  // H col == 4*lane uniformly
  atomicAdd(m + 0, ex * hv.x);
  atomicAdd(m + 1, ex * hv.y);
  atomicAdd(m + 2, ex * hv.z);
  atomicAdd(m + 3, ex * hv.w);
}

// ---------- attention GEMM: A = [mixU/denom | q] (K=256), tanh, panel N=128 ----------
__global__ void attn_gemm_kernel(const float* __restrict__ mixU,
                                 const float* __restrict__ denom,
                                 const float* __restrict__ q,
                                 const _Float16* __restrict__ Bp,  // packed 256x128
                                 float* __restrict__ attn) {
  const int NT = 8;
  extern __shared__ _Float16 smem[];
  stage_b_lds(smem, Bp, 256 * 128);  // 64 KB
  int lane = threadIdx.x & 31;
  int wave = (blockIdx.x * blockDim.x + threadIdx.x) >> 5;
  int nwav = (gridDim.x * blockDim.x) >> 5;
  for (int t = wave; t < (NN >> 4); t += nwav) {
    int row0 = t << 4;
    int m    = row0 + (lane & 15);
    float invd = __builtin_amdgcn_rcpf(denom[m]);
    v8f acc[NT];
#pragma unroll
    for (int nt = 0; nt < NT; ++nt) acc[nt] = (v8f){};
    // phase 1: normalized mix (K = 0..127)
#pragma clang loop unroll(disable)
    for (int k0 = 0; k0 < 128; k0 += 32) {
      const float* p = mixU + (size_t)m * 128 + k0 + ((lane >> 4) << 3);
      v16h a;
#pragma unroll
      for (int i = 0; i < 8; ++i) a[i] = (_Float16)(p[i] * invd);
#pragma unroll
      for (int i = 0; i < 8; ++i) a[8 + i] = (_Float16)(p[16 + i] * invd);
      kstep<NT>(a, smem, k0 >> 5, lane, acc);
    }
    // phase 2: q (K = 128..255)
#pragma clang loop unroll(disable)
    for (int k0 = 0; k0 < 128; k0 += 32) {
      v16h a = load_a_f32(q, 128, row0, k0, lane);
      kstep<NT>(a, smem, (k0 >> 5) + 4, lane, acc);
    }
#pragma unroll
    for (int nt = 0; nt < NT; ++nt)
      store_c<1>(attn, 128, row0, nt << 4, lane, acc[nt], nullptr);
  }
}

// ---------- edge GEMM: A rows gathered [efeat[e] | x[src] | x[dst]] (K=192) ----------
__global__ void edge_gemm_kernel(const float* __restrict__ efeat,
                                 const float* __restrict__ x,
                                 const int* __restrict__ src,
                                 const int* __restrict__ dst,
                                 const _Float16* __restrict__ Bp,  // packed 192x64
                                 const float* __restrict__ bias,
                                 float* __restrict__ out) {
  const int NT = 4;
  extern __shared__ _Float16 smem[];
  stage_b_lds(smem, Bp, 192 * 64);  // 24 KB
  int lane = threadIdx.x & 31;
  int wave = (blockIdx.x * blockDim.x + threadIdx.x) >> 5;
  int nwav = (gridDim.x * blockDim.x) >> 5;
  for (int t = wave; t < (NE >> 4); t += nwav) {
    int row0 = t << 4;
    int e    = row0 + (lane & 15);
    // prefetch next tile's efeat rows (the streaming operand)
    __builtin_prefetch(efeat + (size_t)(e + (nwav << 4)) * D, 0, 1);
    const float* seg0 = efeat + (size_t)e * D;
    const float* seg1 = x + (size_t)src[e] * D;
    const float* seg2 = x + (size_t)dst[e] * D;
    v8f acc[NT];
#pragma unroll
    for (int nt = 0; nt < NT; ++nt) acc[nt] = (v8f){};
#pragma unroll
    for (int seg = 0; seg < 3; ++seg) {
      const float* base = (seg == 0) ? seg0 : (seg == 1) ? seg1 : seg2;
#pragma clang loop unroll(disable)
      for (int kk = 0; kk < 64; kk += 32) {
        const float* p = base + kk + ((lane >> 4) << 3);
        v16h a;
#pragma unroll
        for (int i = 0; i < 8; ++i) a[i] = (_Float16)p[i];
#pragma unroll
        for (int i = 0; i < 8; ++i) a[8 + i] = (_Float16)p[16 + i];
        kstep<NT>(a, smem, (seg << 1) + (kk >> 5), lane, acc);
      }
    }
#pragma unroll
    for (int nt = 0; nt < NT; ++nt)
      store_c<0>(out, 64, row0, nt << 4, lane, acc[nt], bias);
  }
}

// ======================== host-side launcher ========================
extern "C" void kernel_launch(void* const* d_in, const int* in_sizes, int n_in,
                              void* d_out, int out_size, void* d_ws, size_t ws_size,
                              hipStream_t stream) {
  const float* x          = (const float*)d_in[0];
  const float* efeat      = (const float*)d_in[1];
  const int*   src        = (const int*)d_in[2];
  const int*   dst        = (const int*)d_in[3];
  const float* W_att      = (const float*)d_in[4];   // 64 x 128
  const float* W_agg      = (const float*)d_in[5];   // 64 x 64
  const float* W_neis     = (const float*)d_in[6];   // 128 x 64
  const float* W_out_attn = (const float*)d_in[7];   // 256 x 128
  const float* W_node     = (const float*)d_in[8];   // 128 x 64
  const float* b_node     = (const float*)d_in[9];
  const float* W_edge     = (const float*)d_in[10];  // 192 x 64
  const float* b_edge     = (const float*)d_in[11];

  float* node_out = (float*)d_out;                   // [NN x 64]
  float* edge_out = (float*)d_out + (size_t)NN * 64; // [NE x 64]

  // workspace layout (floats); ~158 MB total + 147KB packed weights
  float* ws     = (float*)d_ws;
  float* q      = ws;                      // NN*128
  float* mixU   = ws + 12800000;           // NN*128, reused as `combined`
  float* attn   = ws + 25600000;           // NN*128
  float* scores = ws + 38400000;           // NE
  float* denom  = ws + 39200000;           // NN
  unsigned* maxu = (unsigned*)(ws + 39300000); // NN
  _Float16* hp  = (_Float16*)(ws + 39400000);  // packed f16 weights (32B aligned)
  _Float16* Bp_att  = hp;            // 64*128  = 8192
  _Float16* Bp_agg  = hp + 8192;     // 64*64   = 4096
  _Float16* Bp_neis = hp + 12288;    // 128*64  = 8192
  _Float16* Bp_out  = hp + 20480;    // 256*128 = 32768
  _Float16* Bp_node = hp + 53248;    // 128*64  = 8192
  _Float16* Bp_edge = hp + 61440;    // 192*64  = 12288

  const int TPB = 256;  // 8 waves/block

  // 0) pack all weight matrices into f16 WMMA B-fragment layout
  pack_b_kernel<<<(64 * 128 + TPB - 1) / TPB, TPB, 0, stream>>>(W_att, Bp_att, 64, 128);
  pack_b_kernel<<<(64 * 64 + TPB - 1) / TPB, TPB, 0, stream>>>(W_agg, Bp_agg, 64, 64);
  pack_b_kernel<<<(128 * 64 + TPB - 1) / TPB, TPB, 0, stream>>>(W_neis, Bp_neis, 128, 64);
  pack_b_kernel<<<(256 * 128 + TPB - 1) / TPB, TPB, 0, stream>>>(W_out_attn, Bp_out, 256, 128);
  pack_b_kernel<<<(128 * 64 + TPB - 1) / TPB, TPB, 0, stream>>>(W_node, Bp_node, 128, 64);
  pack_b_kernel<<<(192 * 64 + TPB - 1) / TPB, TPB, 0, stream>>>(W_edge, Bp_edge, 192, 64);

  // 1) zero accumulators (ws is poisoned; must re-init every call)
  init_kernel<<<(NN * 128 + TPB - 1) / TPB, TPB, 0, stream>>>(denom, maxu, mixU);

  // 2) q = x @ W_att  [NN x 128], panel NT=8, LDS B = 64*8*32 = 16 KB
  gemm_kernel<0, 8><<<782, TPB, 64 * 8 * 32, stream>>>(x, 64, Bp_att, nullptr,
                                                       q, 128, NN, 64);

  // 3) per-edge scores + segment max (one wave per edge)
  score_kernel<<<NE / 8, TPB, 0, stream>>>(q, x, efeat, dst, scores, maxu);

  // 4) exp + accumulate denom / unnormalized mix
  softmax_accum_kernel<<<NE / 8, TPB, 0, stream>>>(scores, maxu, x, efeat, dst,
                                                   denom, mixU);

  // 5) attn = tanh([mix | q] @ W_out_attn), LDS B = 64 KB
  attn_gemm_kernel<<<782, TPB, 256 * 128 * 2, stream>>>(mixU, denom, q, Bp_out, attn);

  // 6) combined[:,64:128] = relu(attn @ W_neis), LDS B = 16 KB
  gemm_kernel<2, 4><<<782, TPB, 128 * 4 * 32, stream>>>(attn, 128, Bp_neis, nullptr,
                                                        mixU + 64, 128, NN, 128);

  // 7) combined[:,0:64] = x @ W_agg, LDS B = 8 KB
  gemm_kernel<0, 4><<<782, TPB, 64 * 4 * 32, stream>>>(x, 64, Bp_agg, nullptr,
                                                       mixU, 128, NN, 64);

  // 8) node_out = combined @ W_node + b_node, LDS B = 16 KB
  gemm_kernel<0, 4><<<782, TPB, 128 * 4 * 32, stream>>>(mixU, 128, Bp_node, b_node,
                                                        node_out, 64, NN, 128);

  // 9) edge_out = [efeat | x[src] | x[dst]] @ W_edge + b_edge, LDS B = 24 KB
  edge_gemm_kernel<<<6250, TPB, 192 * 64 * 2, stream>>>(efeat, x, src, dst,
                                                        Bp_edge, b_edge, edge_out);
}
